// EfficientDet_65292092833839
// MI455X (gfx1250) — compile-verified, matrix operand order checked by reading, not run on hardware
//
#include <hip/hip_runtime.h>
#include <hip/hip_bf16.h>

// ---------------------------------------------------------------------------
// Types / helpers
// ---------------------------------------------------------------------------
typedef __attribute__((ext_vector_type(8)))  float  v8f;
typedef __attribute__((ext_vector_type(16))) __bf16 v16bf;
typedef __attribute__((ext_vector_type(4)))  unsigned int v4u;
typedef __attribute__((ext_vector_type(8)))  int v8i;
typedef __attribute__((ext_vector_type(4)))  int v4i;

static __device__ __forceinline__ __bf16 f2bf(float f) {
  union { float f; unsigned u; } v; v.f = f;
  unsigned r = (v.u + 0x7fffu + ((v.u >> 16) & 1u)) >> 16;
  union { unsigned short s; __bf16 b; } o; o.s = (unsigned short)r;
  return o.b;
}
static __device__ __forceinline__ __bf16 bf_zero() {
  union { unsigned short s; __bf16 b; } o; o.s = 0; return o.b;
}
static __device__ __forceinline__ int lane_id() { return (int)(threadIdx.x & 31u); }

// A fragment: 16x32 bf16 from row-major W[ld] (generic pointer: global or LDS)
static __device__ __forceinline__ v16bf frag_a(const __bf16* W, int ld, int m0, int k0) {
  const int l = lane_id();
  const int m = m0 + (l & 15);
  const int kh = (l >> 4) << 3;  // 0 or 8
  const __bf16* p = W + m * ld + k0 + kh;
  v16bf a;
#pragma unroll
  for (int i = 0; i < 8; ++i) { a[i] = p[i]; a[8 + i] = p[16 + i]; }
  return a;
}

// B fragment from pixel-major activations X[P][64]: column n = pixel, k = channel.
// Per lane: two contiguous 16B chunks -> vectorizable global loads.
static __device__ __forceinline__ v16bf frag_b_px(const __bf16* __restrict__ X,
                                                  int n0, int k0) {
  const int l = lane_id();
  const int n = n0 + (l & 15);
  const __bf16* p = X + n * 64 + k0 + ((l >> 4) << 3);
  v16bf b;
#pragma unroll
  for (int i = 0; i < 8; ++i) { b[i] = p[i]; b[8 + i] = p[16 + i]; }
  return b;
}

// Tap-shifted B fragment (3x3 implicit GEMM), pixel-major X[P][64].
// Loads unconditionally from a safe fallback pixel, zeros after (keeps b128 loads).
static __device__ __forceinline__ v16bf frag_b_tap_px(const __bf16* __restrict__ X, int H,
                                                      int n0, int k0, int dy, int dx) {
  const int l = lane_id();
  const int p = n0 + (l & 15);
  const int HW = H * H;
  const int b = p / HW;
  const int r = p - b * HW;
  const int y = r / H;
  const int x = r - y * H;
  const int yy = y + dy, xx = x + dx;
  const bool ok = (yy >= 0) && (yy < H) && (xx >= 0) && (xx < H);
  const int pix = ok ? (b * HW + yy * H + xx) : p;  // fallback: in-bounds pixel
  const __bf16* q = X + pix * 64 + k0 + ((l >> 4) << 3);
  const __bf16 z = bf_zero();
  v16bf v;
#pragma unroll
  for (int i = 0; i < 8; ++i) { v[i] = q[i]; v[8 + i] = q[16 + i]; }
#pragma unroll
  for (int i = 0; i < 16; ++i) v[i] = ok ? v[i] : z;
  return v;
}

// ---------------------------------------------------------------------------
// Weight conversion kernels
// ---------------------------------------------------------------------------
__global__ void cvtbf_kernel(const float* __restrict__ X, __bf16* __restrict__ Y, int n) {
  int t = blockIdx.x * blockDim.x + threadIdx.x;
  if (t < n) Y[t] = f2bf(X[t]);
}

// OIHW (Csrc,64,3,3) -> tap-major bf16 [9][Cpad][64], zero-padded rows
__global__ void wcvt_head_kernel(const float* __restrict__ src, __bf16* __restrict__ dst,
                                 int Csrc, int Cpad) {
  int t = blockIdx.x * blockDim.x + threadIdx.x;
  const int tot = 9 * Cpad * 64;
  if (t >= tot) return;
  const int tap = t / (Cpad * 64);
  const int r = t - tap * (Cpad * 64);
  const int co = r / 64, ci = r - (r / 64) * 64;
  const float v = (co < Csrc) ? src[(co * 64 + ci) * 9 + tap] : 0.0f;
  dst[t] = f2bf(v);
}

// ---------------------------------------------------------------------------
// Projection: NCHW fp32 input -> pixel-major fp32 [B*H*W][64]
// ---------------------------------------------------------------------------
__global__ void proj_kernel(const float* __restrict__ X, const float* __restrict__ W,
                            const float* __restrict__ bias, float* __restrict__ F,
                            int cin, int HW, int P) {
  int t = blockIdx.x * blockDim.x + threadIdx.x;
  if (t >= 64 * P) return;
  const int co = t & 63;
  const int n = t >> 6;
  const int b = n / HW;
  const int p = n - b * HW;
  const float* xp = X + (size_t)b * cin * HW + p;
  const float* wp = W + co * cin;
  float acc = bias[co];
  for (int ci = 0; ci < cin; ++ci) acc += wp[ci] * xp[ci * HW];
  F[t] = acc;
}

// ---------------------------------------------------------------------------
// Fast-normalized fusion (pixel-major fp32 [P][64])
// ---------------------------------------------------------------------------
__global__ void fuse2_up_kernel(const float* __restrict__ A, const float* __restrict__ Bs,
                                const float* __restrict__ wv, float* __restrict__ O,
                                int H, int P) {
  int t = blockIdx.x * blockDim.x + threadIdx.x;
  if (t >= 64 * P) return;
  float w0 = fmaxf(wv[0], 0.0f), w1 = fmaxf(wv[1], 0.0f);
  const float inv = 1.0f / (w0 + w1 + 1e-4f);
  const int c = t & 63;
  const int n = t >> 6;
  const int HW = H * H;
  const int b = n / HW;
  const int r = n - b * HW;
  const int y = r / H, x = r - (r / H) * H;
  const int Hs = H >> 1;
  const float bsv = Bs[(b * Hs * Hs + (y >> 1) * Hs + (x >> 1)) * 64 + c];
  O[t] = (w0 * A[t] + w1 * bsv) * inv;
}

__global__ void fuse2_down_kernel(const float* __restrict__ A, const float* __restrict__ Bb,
                                  const float* __restrict__ wv, float* __restrict__ O,
                                  int H, int P) {
  int t = blockIdx.x * blockDim.x + threadIdx.x;
  if (t >= 64 * P) return;
  float w0 = fmaxf(wv[0], 0.0f), w1 = fmaxf(wv[1], 0.0f);
  const float inv = 1.0f / (w0 + w1 + 1e-4f);
  const int c = t & 63;
  const int n = t >> 6;
  const int HW = H * H;
  const int b = n / HW;
  const int r = n - b * HW;
  const int y = r / H, x = r - (r / H) * H;
  const int Hb = H << 1;
  const float* bp = Bb + (size_t)b * Hb * Hb * 64 + c;
  const int y2 = y << 1, x2 = x << 1;
  const float m = fmaxf(fmaxf(bp[(y2 * Hb + x2) * 64], bp[(y2 * Hb + x2 + 1) * 64]),
                        fmaxf(bp[((y2 + 1) * Hb + x2) * 64], bp[((y2 + 1) * Hb + x2 + 1) * 64]));
  O[t] = (w0 * A[t] + w1 * m) * inv;
}

__global__ void fuse3_kernel(const float* __restrict__ A, const float* __restrict__ Bs,
                             const float* __restrict__ Cb, const float* __restrict__ wv,
                             float* __restrict__ O, int H, int P) {
  int t = blockIdx.x * blockDim.x + threadIdx.x;
  if (t >= 64 * P) return;
  float w0 = fmaxf(wv[0], 0.0f), w1 = fmaxf(wv[1], 0.0f), w2 = fmaxf(wv[2], 0.0f);
  const float inv = 1.0f / (w0 + w1 + w2 + 1e-4f);
  const int c = t & 63;
  const int n = t >> 6;
  const int HW = H * H;
  const int b = n / HW;
  const int r = n - b * HW;
  const int y = r / H, x = r - (r / H) * H;
  const int Hb = H << 1;
  const float* cp = Cb + (size_t)b * Hb * Hb * 64 + c;
  const int y2 = y << 1, x2 = x << 1;
  const float m = fmaxf(fmaxf(cp[(y2 * Hb + x2) * 64], cp[(y2 * Hb + x2 + 1) * 64]),
                        fmaxf(cp[((y2 + 1) * Hb + x2) * 64], cp[((y2 + 1) * Hb + x2 + 1) * 64]));
  O[t] = (w0 * A[t] + w1 * Bs[t] + w2 * m) * inv;
}

// ---------------------------------------------------------------------------
// Depthwise 3x3 (fp32 in -> bf16 out, pixel-major)
// ---------------------------------------------------------------------------
__global__ void dw3_kernel(const float* __restrict__ X, const float* __restrict__ Wd,
                           const float* __restrict__ bd, __bf16* __restrict__ Y,
                           int H, int P) {
  int t = blockIdx.x * blockDim.x + threadIdx.x;
  if (t >= 64 * P) return;
  const int c = t & 63;
  const int n = t >> 6;
  const int HW = H * H;
  const int b = n / HW;
  const int r = n - b * HW;
  const int y = r / H, x = r - (r / H) * H;
  const float* xp = X + (size_t)b * HW * 64 + c;
  const float* w = Wd + c * 9;
  float acc = bd[c];
#pragma unroll
  for (int ky = 0; ky < 3; ++ky) {
#pragma unroll
    for (int kx = 0; kx < 3; ++kx) {
      const int yy = y + ky - 1, xx = x + kx - 1;
      if (yy >= 0 && yy < H && xx >= 0 && xx < H)
        acc += w[ky * 3 + kx] * xp[(yy * H + xx) * 64];
    }
  }
  Y[t] = f2bf(acc);
}

// ---------------------------------------------------------------------------
// Pointwise 1x1 (64x64) GEMM + bias + BN + ReLU via WMMA bf16
// wave -> 64(co) x 16(pix) tile; contiguous fragment loads; contiguous stores
// ---------------------------------------------------------------------------
__global__ __launch_bounds__(256) void pw_bn_relu_kernel(
    const __bf16* __restrict__ X, const __bf16* __restrict__ W,
    const float* __restrict__ pwb, const float* __restrict__ g,
    const float* __restrict__ bb, const float* __restrict__ mean,
    const float* __restrict__ var, float* __restrict__ Y, int P) {
  const int wid = blockIdx.x * 8 + (int)(threadIdx.x >> 5);
  const int n0 = wid * 16;
  if (n0 >= P) return;
  __builtin_prefetch(W, 0, 0);  // gfx1250 global_prefetch_b8
  const v16bf b0 = frag_b_px(X, n0, 0);
  const v16bf b1 = frag_b_px(X, n0, 32);
  const int l = lane_id();
  const int n = n0 + (l & 15);
  const int mh = (l >> 4) << 3;
#pragma unroll
  for (int mt = 0; mt < 4; ++mt) {
    const v16bf a0 = frag_a(W, 64, mt * 16, 0);
    const v16bf a1 = frag_a(W, 64, mt * 16, 32);
    v8f acc;
#pragma unroll
    for (int i = 0; i < 8; ++i) acc[i] = 0.0f;
    acc = __builtin_amdgcn_wmma_f32_16x16x32_bf16(false, a0, false, b0, (short)0, acc,
                                                  false, false);
    acc = __builtin_amdgcn_wmma_f32_16x16x32_bf16(false, a1, false, b1, (short)0, acc,
                                                  false, false);
#pragma unroll
    for (int r = 0; r < 8; ++r) {
      const int co = mt * 16 + mh + r;
      float v = acc[r] + pwb[co];
      v = (v - mean[co]) * rsqrtf(var[co] + 4e-5f) * g[co] + bb[co];
      Y[n * 64 + co] = fmaxf(v, 0.0f);  // 8 consecutive channels per lane
    }
  }
}

// ---------------------------------------------------------------------------
// 3x3 conv over 64 input ch as 9 tap-shifted WMMA GEMMs.
// Weights for this block's co-tile (16 rows x 64 ci x 9 taps = 18KB) are staged
// into LDS once per block via the Tensor Data Mover, then A-fragments come from
// LDS. grid.y = co tile; wave -> 16(co) x 16(pix); 18 WMMAs per wave.
// ---------------------------------------------------------------------------
__global__ __launch_bounds__(256) void head3_kernel(
    const __bf16* __restrict__ X, const __bf16* __restrict__ Wt,
    const float* __restrict__ bias, __bf16* __restrict__ Ybf,
    float* __restrict__ Yf, int H, int P, int Cpad, int Csrc, int relu_out) {
  __shared__ __align__(128) __bf16 wlds[9 * 16 * 64];
  const int mt = blockIdx.y;

#if __has_builtin(__builtin_amdgcn_tensor_load_to_lds)
  if (threadIdx.x < 32) {
    // D# group0: count=1, lds_addr, 57-bit global addr of tile start, type=2
    const unsigned long long ga = (unsigned long long)(const void*)(Wt + (size_t)mt * 1024);
    const unsigned lds = (unsigned)(unsigned long long)(const void*)&wlds[0];
    v4u g0;
    g0[0] = 1u;                                           // count=1
    g0[1] = lds;                                          // lds_addr (bytes)
    g0[2] = (unsigned)ga;                                 // global_addr[31:0]
    g0[3] = ((unsigned)(ga >> 32) & 0x01ffffffu) | 0x80000000u;  // addr[56:32] | type=2
    // D# group1: data_size=2B; 2-D tensor: dim0=1024 contiguous bf16, dim1=9 taps
    v8i g1;
    g1[0] = 0x00010000;                // workgroup_mask=0, data_size=1 (2 bytes)
    g1[1] = (int)(1024u << 16);        // tensor_dim0[15:0]=1024
    g1[2] = (int)(9u << 16);           // tensor_dim0[31:16]=0, tensor_dim1[15:0]=9
    g1[3] = (int)(1024u << 16);        // tensor_dim1 hi=0, tile_dim0=1024
    g1[4] = 9;                         // tile_dim1=9, tile_dim2=0
    g1[5] = Cpad * 64;                 // tensor_dim0_stride (elements)
    g1[6] = 0;
    g1[7] = 0;
    v4i z4; z4[0] = 0; z4[1] = 0; z4[2] = 0; z4[3] = 0;
#if __clang_major__ >= 23
    v8i z8; z8[0]=0; z8[1]=0; z8[2]=0; z8[3]=0; z8[4]=0; z8[5]=0; z8[6]=0; z8[7]=0;
    __builtin_amdgcn_tensor_load_to_lds(g0, g1, z4, z4, z8, 0);
#else
    __builtin_amdgcn_tensor_load_to_lds(g0, g1, z4, z4, 0);
#endif
    __builtin_amdgcn_s_wait_tensorcnt(0);
  }
  __syncthreads();
#else
  for (int i = (int)threadIdx.x; i < 9 * 16 * 64; i += 256)
    wlds[i] = Wt[(i >> 10) * (Cpad * 64) + mt * 1024 + (i & 1023)];
  __syncthreads();
#endif

  const int wid = blockIdx.x * 8 + (int)(threadIdx.x >> 5);
  const int n0 = wid * 16;
  if (n0 >= P) return;
  v8f acc;
#pragma unroll
  for (int i = 0; i < 8; ++i) acc[i] = 0.0f;
#pragma unroll
  for (int tap = 0; tap < 9; ++tap) {
    const int dy = tap / 3 - 1;
    const int dx = tap - (tap / 3) * 3 - 1;
    const v16bf bb0 = frag_b_tap_px(X, H, n0, 0, dy, dx);
    const v16bf bb1 = frag_b_tap_px(X, H, n0, 32, dy, dx);
    const v16bf a0 = frag_a(wlds + tap * 1024, 64, 0, 0);
    const v16bf a1 = frag_a(wlds + tap * 1024, 64, 0, 32);
    acc = __builtin_amdgcn_wmma_f32_16x16x32_bf16(false, a0, false, bb0, (short)0, acc,
                                                  false, false);
    acc = __builtin_amdgcn_wmma_f32_16x16x32_bf16(false, a1, false, bb1, (short)0, acc,
                                                  false, false);
  }
  const int l = lane_id();
  const int n = n0 + (l & 15);
  const int mh = (l >> 4) << 3;
#pragma unroll
  for (int r = 0; r < 8; ++r) {
    const int co = mt * 16 + mh + r;
    float v = acc[r] + ((co < Csrc) ? bias[co] : 0.0f);
    if (relu_out)
      Ybf[n * 64 + co] = f2bf(fmaxf(v, 0.0f));   // contiguous 8 bf16 per lane
    else
      Yf[n * Cpad + co] = v;                     // contiguous 8 f32 per lane
  }
}

// ---------------------------------------------------------------------------
// Anchor decode (reg head fp32, pixel-major [P][48]) -> boxes [8][NT][4]
// ---------------------------------------------------------------------------
__global__ void decode_kernel(const float* __restrict__ Yr, float* __restrict__ boxes,
                              int H, int P, int offL, float stride, int NT) {
  int t = blockIdx.x * blockDim.x + threadIdx.x;
  const int HW = H * H;
  const int tot = 8 * HW * 9;
  if (t >= tot) return;
  const int a = t % 9;
  const int q = t / 9;
  const int hw = q % HW;
  const int b = q / HW;
  const int y = hw / H, x = hw - y * H;
  const int si = a / 3, ri = a - si * 3;
  const float scl = (si == 0) ? 1.0f : ((si == 1) ? 1.2599210498948732f : 1.5874010519681994f);
  const float ratio = (ri == 0) ? 0.5f : ((ri == 1) ? 1.0f : 2.0f);
  const float base = 4.0f * stride;
  const float wa = base * scl * sqrtf(1.0f / ratio);
  const float ha = base * scl * sqrtf(ratio);
  const float cxa = (x + 0.5f) * stride;
  const float cya = (y + 0.5f) * stride;
  const float* rp = Yr + (size_t)(b * HW + hw) * 48 + a * 4;
  const float r0 = rp[0], r1 = rp[1], r2 = rp[2], r3 = rp[3];
  const float cx = cxa + r0 * 0.1f * wa;
  const float cy = cya + r1 * 0.1f * ha;
  const float bw = expf(r2 * 0.2f) * wa;
  const float bh = expf(r3 * 0.2f) * ha;
  const float x1 = fminf(fmaxf(cx - 0.5f * bw, 0.0f), 512.0f);
  const float y1 = fminf(fmaxf(cy - 0.5f * bh, 0.0f), 512.0f);
  const float x2 = fminf(fmaxf(cx + 0.5f * bw, 0.0f), 512.0f);
  const float y2 = fminf(fmaxf(cy + 0.5f * bh, 0.0f), 512.0f);
  const int o = (b * NT + offL + hw * 9 + a) * 4;
  boxes[o + 0] = x1; boxes[o + 1] = y1; boxes[o + 2] = x2; boxes[o + 3] = y2;
}

// cls head fp32 pixel-major [P][192] -> per-anchor max sigmoid score + argmax class
__global__ void clsgather_kernel(const float* __restrict__ Yc, float* __restrict__ scores,
                                 float* __restrict__ classes, int H, int P, int offL, int NT) {
  int t = blockIdx.x * blockDim.x + threadIdx.x;
  const int HW = H * H;
  const int tot = 8 * HW * 9;
  if (t >= tot) return;
  const int a = t % 9;
  const int q = t / 9;
  const int hw = q % HW;
  const int b = q / HW;
  const float* cp = Yc + (size_t)(b * HW + hw) * 192 + a * 20;
  float best = -1e30f; int bi = 0;
#pragma unroll
  for (int k = 0; k < 20; ++k) {
    const float v = cp[k];
    if (v > best) { best = v; bi = k; }
  }
  const int o = b * NT + offL + hw * 9 + a;
  scores[o] = 1.0f / (1.0f + expf(-best));
  classes[o] = (float)bi;
}

// ---------------------------------------------------------------------------
// NMS: one block per image, 100 sequential picks, parallel argmax + suppress
// ---------------------------------------------------------------------------
__global__ __launch_bounds__(256) void nms_kernel(const float* __restrict__ boxes,
                                                  const float* __restrict__ scores,
                                                  const float* __restrict__ classes,
                                                  float* __restrict__ out,
                                                  float* __restrict__ scbuf, int NT) {
  const int img = blockIdx.x;
  const int tid = threadIdx.x;
  const float* bx = boxes + (size_t)img * NT * 4;
  float* sc = scbuf + (size_t)img * NT;
  for (int i = tid; i < NT; i += 256) {
    const float s = scores[(size_t)img * NT + i];
    sc[i] = (s > 0.05f) ? s : -1.0f;
  }
  __syncthreads();
  __shared__ float smax[256];
  __shared__ int sidx[256];
  for (int det = 0; det < 100; ++det) {
    float m = -1e30f; int mi = 0;
    for (int i = tid; i < NT; i += 256)
      if (sc[i] > m) { m = sc[i]; mi = i; }
    smax[tid] = m; sidx[tid] = mi;
    __syncthreads();
    for (int s = 128; s > 0; s >>= 1) {
      if (tid < s && smax[tid + s] > smax[tid]) {
        smax[tid] = smax[tid + s];
        sidx[tid] = sidx[tid + s];
      }
      __syncthreads();
    }
    const int i = sidx[0];
    const float s = smax[0];
    const float b0 = bx[i * 4 + 0], b1 = bx[i * 4 + 1];
    const float b2 = bx[i * 4 + 2], b3 = bx[i * 4 + 3];
    const float valid = (s > 0.0f) ? 1.0f : 0.0f;
    if (tid == 0) {
      float* row = out + (size_t)(img * 100 + det) * 6;
      row[0] = b0 * valid; row[1] = b1 * valid; row[2] = b2 * valid; row[3] = b3 * valid;
      row[4] = s * valid;
      row[5] = classes[(size_t)img * NT + i] * valid;
    }
    const float areaI = (b2 - b0) * (b3 - b1);
    for (int j = tid; j < NT; j += 256) {
      const float jx1 = bx[j * 4 + 0], jy1 = bx[j * 4 + 1];
      const float jx2 = bx[j * 4 + 2], jy2 = bx[j * 4 + 3];
      const float ix1 = fmaxf(b0, jx1), iy1 = fmaxf(b1, jy1);
      const float ix2 = fminf(b2, jx2), iy2 = fminf(b3, jy2);
      const float inter = fmaxf(ix2 - ix1, 0.0f) * fmaxf(iy2 - iy1, 0.0f);
      const float areaJ = (jx2 - jx1) * (jy2 - jy1);
      const float iou = inter / (areaI + areaJ - inter + 1e-8f);
      if (iou > 0.5f || j == i) sc[j] = -1.0f;
    }
    __syncthreads();
  }
}

// ---------------------------------------------------------------------------
// Host orchestration
// ---------------------------------------------------------------------------
static inline int cdiv(int a, int b) { return (a + b - 1) / b; }

// Leaf indices assuming jax tree order (dict keys sorted, lists in order):
//  0..4  : p3..p7 input feature maps
//  5..220: bifpn 3 layers x 72 leaves
//          each layer: 8 conv blocks (sorted: conv3_up, conv4_down, conv4_up,
//          conv5_down, conv5_up, conv6_down, conv6_up, conv7_down), each block
//          8 leaves (bn_b, bn_g, bn_mean, bn_var, dw_b, dw_w, pw_b, pw_w),
//          then fuse weights p3_w1,p4_w1,p4_w2,p5_w1,p5_w2,p6_w1,p6_w2,p7_w2
//  221..228: cls head (head_b, head_w, l0_w, l0_b, l1_w, l1_b, l2_w, l2_b)
//  229..238: proj0_b, proj0_w, ..., proj4_b, proj4_w
//  239..246: reg head (same layout as cls)
#define BIFPN_BASE 5
#define CLS_BASE   221
#define PROJ_BASE  229
#define REG_BASE   239

extern "C" void kernel_launch(void* const* d_in, const int* in_sizes, int n_in,
                              void* d_out, int out_size, void* d_ws, size_t ws_size,
                              hipStream_t stream) {
  (void)in_sizes; (void)n_in; (void)out_size; (void)ws_size;
  const int HH[5] = {64, 32, 16, 8, 4};
  const int PP[5] = {8 * 64 * 64, 8 * 32 * 32, 8 * 16 * 16, 8 * 8 * 8, 8 * 4 * 4};
  const int CIN[5] = {40, 80, 112, 192, 320};
  const float STR[5] = {8.f, 16.f, 32.f, 64.f, 128.f};
  const int AOFF[5] = {0, 36864, 46080, 48384, 48960};
  const int NT = 49104;

  auto F32IN = [&](int i) -> const float* { return (const float*)d_in[i]; };

  // ---- workspace carve-out ----
  char* wsb = (char*)d_ws;
  size_t cur = 0;
  auto alloc = [&](size_t bytes) -> char* {
    cur = (cur + 255) & ~(size_t)255;
    char* p = wsb + cur;
    cur += bytes;
    return p;
  };
  float* featA[5]; float* featB[5]; __bf16* featbf[5];
  for (int l = 0; l < 5; ++l) featA[l] = (float*)alloc((size_t)64 * PP[l] * 4);
  for (int l = 0; l < 5; ++l) featB[l] = (float*)alloc((size_t)64 * PP[l] * 4);
  for (int l = 0; l < 5; ++l) featbf[l] = (__bf16*)alloc((size_t)64 * PP[l] * 2);
  float* u4 = (float*)alloc((size_t)64 * PP[1] * 4);
  float* u5 = (float*)alloc((size_t)64 * PP[2] * 4);
  float* u6 = (float*)alloc((size_t)64 * PP[3] * 4);
  float* tfuse = (float*)alloc((size_t)64 * PP[0] * 4);
  __bf16* tdw = (__bf16*)alloc((size_t)64 * PP[0] * 2);
  __bf16* pwW = (__bf16*)alloc((size_t)24 * 4096 * 2);
  __bf16* hlwC = (__bf16*)alloc((size_t)3 * 9 * 64 * 64 * 2);
  __bf16* hlwR = (__bf16*)alloc((size_t)3 * 9 * 64 * 64 * 2);
  __bf16* hfwC = (__bf16*)alloc((size_t)9 * 192 * 64 * 2);
  __bf16* hfwR = (__bf16*)alloc((size_t)9 * 48 * 64 * 2);
  __bf16* hbA = (__bf16*)alloc((size_t)64 * PP[0] * 2);
  __bf16* hbB = (__bf16*)alloc((size_t)64 * PP[0] * 2);
  float* hout = (float*)alloc((size_t)192 * PP[0] * 4);
  float* boxes = (float*)alloc((size_t)8 * NT * 4 * 4);
  float* scores = (float*)alloc((size_t)8 * NT * 4);
  float* classes = (float*)alloc((size_t)8 * NT * 4);
  float* scbuf = (float*)alloc((size_t)8 * NT * 4);

  // ---- weight conversion (fp32 -> bf16, reorder 3x3 to tap-major) ----
  for (int L = 0; L < 3; ++L)
    for (int blk = 0; blk < 8; ++blk)
      cvtbf_kernel<<<cdiv(4096, 256), 256, 0, stream>>>(
          F32IN(BIFPN_BASE + L * 72 + blk * 8 + 7), pwW + (L * 8 + blk) * 4096, 4096);
  for (int i = 0; i < 3; ++i) {
    wcvt_head_kernel<<<cdiv(9 * 64 * 64, 256), 256, 0, stream>>>(
        F32IN(CLS_BASE + 2 + 2 * i), hlwC + i * 9 * 64 * 64, 64, 64);
    wcvt_head_kernel<<<cdiv(9 * 64 * 64, 256), 256, 0, stream>>>(
        F32IN(REG_BASE + 2 + 2 * i), hlwR + i * 9 * 64 * 64, 64, 64);
  }
  wcvt_head_kernel<<<cdiv(9 * 192 * 64, 256), 256, 0, stream>>>(F32IN(CLS_BASE + 1), hfwC, 180, 192);
  wcvt_head_kernel<<<cdiv(9 * 48 * 64, 256), 256, 0, stream>>>(F32IN(REG_BASE + 1), hfwR, 36, 48);

  // ---- projections: NCHW -> pixel-major fp32 ----
  for (int l = 0; l < 5; ++l)
    proj_kernel<<<cdiv(64 * PP[l], 256), 256, 0, stream>>>(
        F32IN(l), F32IN(PROJ_BASE + l * 2 + 1), F32IN(PROJ_BASE + l * 2 + 0),
        featA[l], CIN[l], HH[l] * HH[l], PP[l]);

  // ---- BiFPN ----
  auto conv_block = [&](int L, int blk, const float* src, float* dst, int lvl) {
    const int base = BIFPN_BASE + L * 72 + blk * 8;
    const int H = HH[lvl], P = PP[lvl];
    dw3_kernel<<<cdiv(64 * P, 256), 256, 0, stream>>>(src, F32IN(base + 5), F32IN(base + 4),
                                                      tdw, H, P);
    pw_bn_relu_kernel<<<cdiv(P / 16, 8), 256, 0, stream>>>(
        tdw, pwW + (L * 8 + blk) * 4096, F32IN(base + 6), F32IN(base + 1), F32IN(base + 0),
        F32IN(base + 2), F32IN(base + 3), dst, P);
  };

  for (int L = 0; L < 3; ++L) {
    float** fin = (L % 2 == 0) ? featA : featB;
    float** fout = (L % 2 == 0) ? featB : featA;
    const int fw = BIFPN_BASE + L * 72 + 64;  // fuse weights base
    // top-down
    fuse2_up_kernel<<<cdiv(64 * PP[3], 256), 256, 0, stream>>>(fin[3], fin[4], F32IN(fw + 5),
                                                               tfuse, HH[3], PP[3]);
    conv_block(L, 6, tfuse, u6, 3);                       // conv6_up -> p6u
    fuse2_up_kernel<<<cdiv(64 * PP[2], 256), 256, 0, stream>>>(fin[2], u6, F32IN(fw + 3),
                                                               tfuse, HH[2], PP[2]);
    conv_block(L, 4, tfuse, u5, 2);                       // conv5_up -> p5u
    fuse2_up_kernel<<<cdiv(64 * PP[1], 256), 256, 0, stream>>>(fin[1], u5, F32IN(fw + 1),
                                                               tfuse, HH[1], PP[1]);
    conv_block(L, 2, tfuse, u4, 1);                       // conv4_up -> p4u
    fuse2_up_kernel<<<cdiv(64 * PP[0], 256), 256, 0, stream>>>(fin[0], u4, F32IN(fw + 0),
                                                               tfuse, HH[0], PP[0]);
    conv_block(L, 0, tfuse, fout[0], 0);                  // conv3_up -> p3o
    // bottom-up
    fuse3_kernel<<<cdiv(64 * PP[1], 256), 256, 0, stream>>>(fin[1], u4, fout[0], F32IN(fw + 2),
                                                            tfuse, HH[1], PP[1]);
    conv_block(L, 1, tfuse, fout[1], 1);                  // conv4_down
    fuse3_kernel<<<cdiv(64 * PP[2], 256), 256, 0, stream>>>(fin[2], u5, fout[1], F32IN(fw + 4),
                                                            tfuse, HH[2], PP[2]);
    conv_block(L, 3, tfuse, fout[2], 2);                  // conv5_down
    fuse3_kernel<<<cdiv(64 * PP[3], 256), 256, 0, stream>>>(fin[3], u6, fout[2], F32IN(fw + 6),
                                                            tfuse, HH[3], PP[3]);
    conv_block(L, 5, tfuse, fout[3], 3);                  // conv6_down
    fuse2_down_kernel<<<cdiv(64 * PP[4], 256), 256, 0, stream>>>(fin[4], fout[3], F32IN(fw + 7),
                                                                 tfuse, HH[4], PP[4]);
    conv_block(L, 7, tfuse, fout[4], 4);                  // conv7_down
  }
  // After 3 layers final features are in featB; convert to bf16 for the heads.
  for (int l = 0; l < 5; ++l)
    cvtbf_kernel<<<cdiv(64 * PP[l], 256), 256, 0, stream>>>(featB[l], featbf[l], 64 * PP[l]);

  // ---- heads + decode/gather ----
  for (int l = 0; l < 5; ++l) {
    const int H = HH[l], P = PP[l];
    dim3 g1(cdiv(P / 16, 8), 4, 1);
    // reg
    head3_kernel<<<g1, 256, 0, stream>>>(featbf[l], hlwR + 0 * 9 * 64 * 64, F32IN(REG_BASE + 3),
                                         hbA, nullptr, H, P, 64, 64, 1);
    head3_kernel<<<g1, 256, 0, stream>>>(hbA, hlwR + 1 * 9 * 64 * 64, F32IN(REG_BASE + 5),
                                         hbB, nullptr, H, P, 64, 64, 1);
    head3_kernel<<<g1, 256, 0, stream>>>(hbB, hlwR + 2 * 9 * 64 * 64, F32IN(REG_BASE + 7),
                                         hbA, nullptr, H, P, 64, 64, 1);
    dim3 g2(cdiv(P / 16, 8), 3, 1);
    head3_kernel<<<g2, 256, 0, stream>>>(hbA, hfwR, F32IN(REG_BASE + 0), nullptr, hout,
                                         H, P, 48, 36, 0);
    decode_kernel<<<cdiv(8 * H * H * 9, 256), 256, 0, stream>>>(hout, boxes, H, P, AOFF[l],
                                                                STR[l], NT);
    // cls
    head3_kernel<<<g1, 256, 0, stream>>>(featbf[l], hlwC + 0 * 9 * 64 * 64, F32IN(CLS_BASE + 3),
                                         hbA, nullptr, H, P, 64, 64, 1);
    head3_kernel<<<g1, 256, 0, stream>>>(hbA, hlwC + 1 * 9 * 64 * 64, F32IN(CLS_BASE + 5),
                                         hbB, nullptr, H, P, 64, 64, 1);
    head3_kernel<<<g1, 256, 0, stream>>>(hbB, hlwC + 2 * 9 * 64 * 64, F32IN(CLS_BASE + 7),
                                         hbA, nullptr, H, P, 64, 64, 1);
    dim3 g3(cdiv(P / 16, 8), 12, 1);
    head3_kernel<<<g3, 256, 0, stream>>>(hbA, hfwC, F32IN(CLS_BASE + 0), nullptr, hout,
                                         H, P, 192, 180, 0);
    clsgather_kernel<<<cdiv(8 * H * H * 9, 256), 256, 0, stream>>>(hout, scores, classes,
                                                                   H, P, AOFF[l], NT);
  }

  // ---- NMS ----
  nms_kernel<<<8, 256, 0, stream>>>(boxes, scores, classes, (float*)d_out, scbuf, NT);
}
